// MultiHeadSelfAttention_996432412827
// MI455X (gfx1250) — matrix operand order repaired and checked by reading, not compiled, not run on hardware
//
#include <hip/hip_runtime.h>
#include <hip/hip_bf16.h>

typedef __bf16 bf16_t;
typedef __attribute__((ext_vector_type(16))) __bf16 v16bf;
typedef __attribute__((ext_vector_type(8)))  __bf16 v8bf;
typedef __attribute__((ext_vector_type(4)))  __bf16 v4bf;
typedef __attribute__((ext_vector_type(8)))  float  v8f;
typedef __attribute__((ext_vector_type(4)))  float  v4f;

#define DIM   1024
#define NH    16
#define HD    64
#define BATCH 2
#define SEQ   2048
#define MTOT  (BATCH * SEQ)   // 4096
#define ATT_SCALE 0.125f      // 1/sqrt(64), exact power of two

static __device__ __forceinline__ v8f wmma_bf16(v16bf a, v16bf b, v8f c) {
    // D = A(16x32 bf16) * B(32x16 bf16) + C(16x16 f32)
    return __builtin_amdgcn_wmma_f32_16x16x32_bf16(false, a, false, b, (short)0, c,
                                                   false, false);
}

// A fragment (16x32 bf16) from row-major src (ld elems/row) at (row0, k0).
// Lane L: row = row0 + (L&15); dwords 0-3 hold K = k0 + half*8 + {0..7} (pairs),
// dwords 4-7 hold K = k0 + half*8 + {16..23}  (half = L>=16).
static __device__ __forceinline__ v16bf load_afrag(const bf16_t* src, int ld,
                                                   int row0, int k0) {
    const int lane = threadIdx.x & 31;
    const int half = lane >> 4;
    const bf16_t* p = src + (size_t)(row0 + (lane & 15)) * ld + k0 + half * 8;
    v8bf lo = *reinterpret_cast<const v8bf*>(p);
    v8bf hi = *reinterpret_cast<const v8bf*>(p + 16);
    v16bf f;
#pragma unroll
    for (int i = 0; i < 8; ++i) { f[i] = lo[i]; f[8 + i] = hi[i]; }
    return f;
}

// B fragment (32x16 bf16) from column-major-B storage bt[N][K] (each output
// column's K values contiguous). Lane L: col = n0 + (L&15); elements 0-7 hold
// K = k0 + half*8 + {0..7}, elements 8-15 hold K = k0 + half*8 + {16..23}.
static __device__ __forceinline__ v16bf load_bfrag(const bf16_t* bt, int ld,
                                                   int n0, int k0) {
    const int lane = threadIdx.x & 31;
    const bf16_t* p = bt + (size_t)(n0 + (lane & 15)) * ld + k0 + (lane >> 4) * 8;
    v8bf lo = *reinterpret_cast<const v8bf*>(p);
    v8bf hi = *reinterpret_cast<const v8bf*>(p + 16);
    v16bf f;
#pragma unroll
    for (int i = 0; i < 8; ++i) { f[i] = lo[i]; f[8 + i] = hi[i]; }
    return f;
}

// ---------------- stage 0: precision conversion ----------------

// 4 elements per thread: b128 load, b64 store
__global__ void f32_to_bf16_kernel(const float* __restrict__ in,
                                   bf16_t* __restrict__ out, int n4) {
    int i = blockIdx.x * blockDim.x + threadIdx.x;
    if (i < n4) {
        v4f x = reinterpret_cast<const v4f*>(in)[i];
        v4bf y;
#pragma unroll
        for (int j = 0; j < 4; ++j) y[j] = (bf16_t)x[j];
        reinterpret_cast<v4bf*>(out)[i] = y;
    }
}

// LDS-tiled transpose: out[n][k] = (bf16)in[k][n], DIM x DIM.
// Coalesced reads and writes; 33-wide LDS rows to avoid bank conflicts.
__global__ void __launch_bounds__(256) transpose_to_bf16_kernel(
        const float* __restrict__ in, bf16_t* __restrict__ out) {
    __shared__ bf16_t tile[32][33];
    const int tx = threadIdx.x & 31;
    const int ty = threadIdx.x >> 5;              // 0..7
    const int k0 = (blockIdx.x & 31) * 32;        // source row tile
    const int n0 = (blockIdx.x >> 5) * 32;        // source col tile
#pragma unroll
    for (int r = 0; r < 4; ++r) {
        const int k = k0 + ty + 8 * r;
        tile[ty + 8 * r][tx] = (bf16_t)in[(size_t)k * DIM + n0 + tx];
    }
    __syncthreads();
#pragma unroll
    for (int r = 0; r < 4; ++r) {
        const int n = n0 + ty + 8 * r;
        out[(size_t)n * DIM + k0 + tx] = tile[tx][ty + 8 * r];
    }
}

// ---------------- stage 1: fused QKV projection ----------------
// C[m, n] = x[m, :] . W[:, n] + bias,  N = 3 * DIM (q | k | v)
// Q (pre-scaled by 1/sqrt(d)), K written [B][H][T][D]; V written [B][H][D][T].

__global__ void __launch_bounds__(256) qkv_gemm_kernel(
        const bf16_t* __restrict__ xb,   // [MTOT][DIM]
        const bf16_t* __restrict__ wt,   // [3][DIM(N)][DIM(K)]
        const float* __restrict__ biasq, const float* __restrict__ biask,
        const float* __restrict__ biasv,
        bf16_t* __restrict__ qo, bf16_t* __restrict__ ko, bf16_t* __restrict__ vo) {
    const int wave = (blockIdx.x * blockDim.x + threadIdx.x) >> 5;
    const int MT = MTOT / 32;                  // 128
    const int tm = wave % MT, tn = wave / MT;  // tn in [0, 96)
    const int row0 = tm * 32;
    const int col0 = tn * 32;                  // global over 3072
    const int wsel = col0 >> 10;               // 0=q 1=k 2=v
    const int ncol = col0 & 1023;
    const bf16_t* bt = wt + (size_t)wsel * DIM * DIM;

    v8f acc[2][2] = {};
    for (int k0 = 0; k0 < DIM; k0 += 32) {
        v16bf a0 = load_afrag(xb, DIM, row0, k0);
        v16bf a1 = load_afrag(xb, DIM, row0 + 16, k0);
        v16bf b0 = load_bfrag(bt, DIM, ncol, k0);
        v16bf b1 = load_bfrag(bt, DIM, ncol + 16, k0);
        acc[0][0] = wmma_bf16(a0, b0, acc[0][0]);
        acc[0][1] = wmma_bf16(a0, b1, acc[0][1]);
        acc[1][0] = wmma_bf16(a1, b0, acc[1][0]);
        acc[1][1] = wmma_bf16(a1, b1, acc[1][1]);
    }

    const float* bias = (wsel == 0) ? biasq : (wsel == 1) ? biask : biasv;
    const int lane = threadIdx.x & 31;
    const int half = lane >> 4;
#pragma unroll
    for (int i = 0; i < 2; ++i) {
#pragma unroll
        for (int j = 0; j < 2; ++j) {
            const int n = ncol + j * 16 + (lane & 15);
            const float bval = bias[n];
            const int h = n >> 6, d = n & 63;
#pragma unroll
            for (int e = 0; e < 8; ++e) {
                const int m = row0 + i * 16 + half * 8 + e;
                const int bb = m >> 11, t = m & 2047;
                float val = acc[i][j][e] + bval;
                if (wsel == 0) val *= ATT_SCALE;   // fold softmax scale into Q (exact)
                if (wsel == 2) {
                    vo[(((size_t)bb * NH + h) * HD + d) * SEQ + t] = (bf16_t)val;
                } else {
                    bf16_t* dst = (wsel == 0) ? qo : ko;
                    dst[(((size_t)bb * NH + h) * SEQ + t) * HD + d] = (bf16_t)val;
                }
            }
        }
    }
}

// ---------------- stage 2: causal flash attention ----------------
// One wave = 16 query rows; block = 8 waves = 128 queries of one (b,h).

__global__ void __launch_bounds__(256) attn_kernel(
        const bf16_t* __restrict__ q,   // [B][H][T][D], pre-scaled
        const bf16_t* __restrict__ kk,  // [B][H][T][D]
        const bf16_t* __restrict__ v,   // [B][H][D][T]
        bf16_t* __restrict__ ctx) {     // [B][T][H*D] == [MTOT][DIM]
    __shared__ __align__(16) bf16_t plds[8][16 * 32];

    const int wid = threadIdx.x >> 5;
    const int lane = threadIdx.x & 31;
    const int half = lane >> 4;
    const int bh = blockIdx.x;          // 0..B*NH-1
    const int b = bh >> 4, h = bh & 15;
    const int qbase = blockIdx.y * 128 + wid * 16;

    const bf16_t* qp = q + (size_t)bh * SEQ * HD;
    const bf16_t* kp = kk + (size_t)bh * SEQ * HD;
    const bf16_t* vp = v + (size_t)bh * HD * SEQ;

    // Q fragments for the whole kernel (head dim 64 = 2 K-steps of 32)
    const v16bf aq0 = load_afrag(qp, HD, qbase, 0);
    const v16bf aq1 = load_afrag(qp, HD, qbase, 32);

    v8f o[4] = {};              // 16 x 64 fp32 ctx accumulator
    float mrow[8], lrow[8];     // per-row online softmax stats (row = e + 8*half)
#pragma unroll
    for (int e = 0; e < 8; ++e) { mrow[e] = -1e30f; lrow[e] = 0.f; }

    const int kend = qbase + 16;        // causal: keys <= qbase+15
    for (int kb = 0; kb < kend; kb += 32) {
        // scores 16x32 as two 16x16 C fragments
        v8f s0 = {}, s1 = {};
        s0 = wmma_bf16(aq0, load_bfrag(kp, HD, kb, 0), s0);
        s0 = wmma_bf16(aq1, load_bfrag(kp, HD, kb, 32), s0);
        s1 = wmma_bf16(aq0, load_bfrag(kp, HD, kb + 16, 0), s1);
        s1 = wmma_bf16(aq1, load_bfrag(kp, HD, kb + 16, 32), s1);

        const int c0 = kb + (lane & 15);
#pragma unroll
        for (int e = 0; e < 8; ++e) {
            const int row = qbase + half * 8 + e;
            if (c0 > row) s0[e] = -1e30f;          // causal mask
            if (c0 + 16 > row) s1[e] = -1e30f;
        }

        // online softmax; rows are spread across the 16 lanes of each half
#pragma unroll
        for (int e = 0; e < 8; ++e) {
            float t = fmaxf(s0[e], s1[e]);
#pragma unroll
            for (int msk = 8; msk >= 1; msk >>= 1)
                t = fmaxf(t, __shfl_xor(t, msk, 32));
            const float mnew = fmaxf(mrow[e], t);
            const float alpha = __expf(mrow[e] - mnew);
            const float p0 = __expf(s0[e] - mnew);
            const float p1 = __expf(s1[e] - mnew);
            float rs = p0 + p1;
#pragma unroll
            for (int msk = 8; msk >= 1; msk >>= 1)
                rs += __shfl_xor(rs, msk, 32);
            lrow[e] = lrow[e] * alpha + rs;
            mrow[e] = mnew;
#pragma unroll
            for (int i = 0; i < 4; ++i) o[i][e] *= alpha;
            // stash P (C-layout -> LDS row-major 16x32)
            const int r = half * 8 + e;
            plds[wid][r * 32 + (lane & 15)] = (bf16_t)p0;
            plds[wid][r * 32 + (lane & 15) + 16] = (bf16_t)p1;
        }
        asm volatile("s_wait_dscnt 0" ::: "memory");  // wave-local LDS RAW

        // P (16x32) x V (32x64): 4 WMMAs, V pre-transposed so K pairs contiguous
        const v16bf ap = load_afrag(&plds[wid][0], 32, 0, 0);
#pragma unroll
        for (int i = 0; i < 4; ++i)
            o[i] = wmma_bf16(ap, load_bfrag(vp, SEQ, i * 16, kb), o[i]);
    }

    // normalize and write ctx as [B][T][H*D]
#pragma unroll
    for (int i = 0; i < 4; ++i) {
        const int d = i * 16 + (lane & 15);
#pragma unroll
        for (int e = 0; e < 8; ++e) {
            const int t = qbase + half * 8 + e;
            const float val = o[i][e] / lrow[e];
            ctx[((size_t)b * SEQ + t) * DIM + h * HD + d] = (bf16_t)val;
        }
    }
}

// ---------------- stage 3: output projection (fp32 out) ----------------

__global__ void __launch_bounds__(256) out_gemm_kernel(
        const bf16_t* __restrict__ ctx,  // [MTOT][DIM]
        const bf16_t* __restrict__ wot,  // [DIM(N)][DIM(K)]
        const float* __restrict__ bo,
        float* __restrict__ out) {       // [MTOT][DIM]
    const int wave = (blockIdx.x * blockDim.x + threadIdx.x) >> 5;
    const int MT = MTOT / 32;
    const int tm = wave % MT, tn = wave / MT;
    const int row0 = tm * 32, col0 = tn * 32;

    v8f acc[2][2] = {};
    for (int k0 = 0; k0 < DIM; k0 += 32) {
        v16bf a0 = load_afrag(ctx, DIM, row0, k0);
        v16bf a1 = load_afrag(ctx, DIM, row0 + 16, k0);
        v16bf b0 = load_bfrag(wot, DIM, col0, k0);
        v16bf b1 = load_bfrag(wot, DIM, col0 + 16, k0);
        acc[0][0] = wmma_bf16(a0, b0, acc[0][0]);
        acc[0][1] = wmma_bf16(a0, b1, acc[0][1]);
        acc[1][0] = wmma_bf16(a1, b0, acc[1][0]);
        acc[1][1] = wmma_bf16(a1, b1, acc[1][1]);
    }

    const int lane = threadIdx.x & 31;
    const int half = lane >> 4;
#pragma unroll
    for (int i = 0; i < 2; ++i) {
#pragma unroll
        for (int j = 0; j < 2; ++j) {
            const int n = col0 + j * 16 + (lane & 15);
            const float bval = bo[n];
#pragma unroll
            for (int e = 0; e < 8; ++e) {
                const int m = row0 + i * 16 + half * 8 + e;
                out[(size_t)m * DIM + n] = acc[i][j][e] + bval;
            }
        }
    }
}

// ---------------- host launch ----------------

extern "C" void kernel_launch(void* const* d_in, const int* in_sizes, int n_in,
                              void* d_out, int out_size, void* d_ws, size_t ws_size,
                              hipStream_t stream) {
    const float* x  = (const float*)d_in[0];
    const float* Wq = (const float*)d_in[1];
    const float* bq = (const float*)d_in[2];
    const float* Wk = (const float*)d_in[3];
    const float* bk = (const float*)d_in[4];
    const float* Wv = (const float*)d_in[5];
    const float* bv = (const float*)d_in[6];
    const float* Wo = (const float*)d_in[7];
    const float* bo = (const float*)d_in[8];
    float* out = (float*)d_out;

    char* ws = (char*)d_ws;
    size_t off = 0;
    auto alloc = [&](size_t bytes) {
        void* p = ws + off;
        off = (off + bytes + 255) & ~(size_t)255;
        return p;
    };
    bf16_t* xb   = (bf16_t*)alloc((size_t)MTOT * DIM * 2);       // x bf16
    bf16_t* wt   = (bf16_t*)alloc((size_t)3 * DIM * DIM * 2);    // Wq^T|Wk^T|Wv^T
    bf16_t* wot  = (bf16_t*)alloc((size_t)DIM * DIM * 2);        // Wo^T
    bf16_t* qb   = (bf16_t*)alloc((size_t)MTOT * DIM * 2);       // [B,H,T,D]
    bf16_t* kbuf = (bf16_t*)alloc((size_t)MTOT * DIM * 2);       // [B,H,T,D]
    bf16_t* vbuf = (bf16_t*)alloc((size_t)MTOT * DIM * 2);       // [B,H,D,T]
    bf16_t* ctxb = (bf16_t*)alloc((size_t)MTOT * DIM * 2);       // [B,T,H*D]
    (void)ws_size; (void)in_sizes; (void)n_in; (void)out_size;

    const int nconv4 = MTOT * DIM / 4;
    f32_to_bf16_kernel<<<(nconv4 + 255) / 256, 256, 0, stream>>>(x, xb, nconv4);
    const int ntb = (DIM / 32) * (DIM / 32);   // 1024 tile-blocks per matrix
    transpose_to_bf16_kernel<<<ntb, 256, 0, stream>>>(Wq, wt);
    transpose_to_bf16_kernel<<<ntb, 256, 0, stream>>>(Wk, wt + (size_t)DIM * DIM);
    transpose_to_bf16_kernel<<<ntb, 256, 0, stream>>>(Wv, wt + (size_t)2 * DIM * DIM);
    transpose_to_bf16_kernel<<<ntb, 256, 0, stream>>>(Wo, wot);

    // QKV: (4096/32) * (3072/32) = 12288 waves / 8 per block
    qkv_gemm_kernel<<<12288 / 8, 256, 0, stream>>>(xb, wt, bq, bk, bv, qb, kbuf, vbuf);

    // attention: one (b,h) per blockIdx.x, 128 queries per blockIdx.y
    attn_kernel<<<dim3(BATCH * NH, SEQ / 128), 256, 0, stream>>>(qb, kbuf, vbuf, ctxb);

    // out projection: (4096/32) * (1024/32) = 4096 waves / 8 per block
    out_gemm_kernel<<<4096 / 8, 256, 0, stream>>>(ctxb, wot, bo, out);
}
